// GraphConv_52673478918720
// MI455X (gfx1250) — matrix operand order repaired and checked by reading, not compiled, not run on hardware
//
#include <hip/hip_runtime.h>

typedef __attribute__((ext_vector_type(16))) _Float16 v16h;
typedef __attribute__((ext_vector_type(8)))  _Float16 v8h;
typedef __attribute__((ext_vector_type(4)))  _Float16 v4h;
typedef __attribute__((ext_vector_type(8)))  float    v8f;

#define D 128              // D_IN == D_OUT == 128
#define D4 (D / 4)         // float4 per row

// ---------------------------------------------------------------------------
// Kernel 0: w [K=128][N=128] f32 (row-major)  ->  wT [N][K] f16
// Gives each WMMA B-fragment lane two contiguous 8-half runs.
// ---------------------------------------------------------------------------
__global__ __launch_bounds__(256) void prep_w_kernel(const float* __restrict__ w,
                                                     _Float16* __restrict__ wT) {
    int idx = blockIdx.x * blockDim.x + threadIdx.x;   // 0..16383
    int n = idx & (D - 1);
    int k = idx >> 7;
    wT[(size_t)n * D + k] = (_Float16)w[(size_t)k * D + n];
}

// ---------------------------------------------------------------------------
// Kernel 1: h16[m][n] = f16( (x[m] ∘ mask[m]) @ w )
// Block = 256 threads (8 waves). Block owns 16 rows; wave w owns cols [16w,16w+16).
// A tile (16x128, f16, dropout applied) staged once in LDS, shared by all waves.
// 4 x v_wmma_f32_16x16x32_f16 per wave (K = 128). Result bounced through the
// same LDS tile so global stores are coalesced 16B row-major writes.
// ---------------------------------------------------------------------------
__global__ __launch_bounds__(256) void gemm_wmma_kernel(const float* __restrict__ x,
                                                        const float* __restrict__ mask,
                                                        const _Float16* __restrict__ wT,
                                                        _Float16* __restrict__ h16,
                                                        int nNodes) {
    __shared__ __align__(16) _Float16 aTile[16 * D];   // 4 KB, reused for output

    const int tid  = threadIdx.x;
    const int row0 = blockIdx.x * 16;

    // ---- stage A = (x ∘ mask) as f16 into LDS (8 elements per thread) ----
    {
        int e = tid * 8;                 // 0..2047
        int r = e >> 7;                  // row within tile
        int k = e & (D - 1);             // col
        int row = row0 + r;
        v8h hv;
        if (row < nNodes) {
            const float4* xp = (const float4*)(x    + (size_t)row * D + k);
            const float4* mp = (const float4*)(mask + (size_t)row * D + k);
            float4 x0 = xp[0], x1 = xp[1];
            float4 m0 = mp[0], m1 = mp[1];
            float xs[8] = {x0.x, x0.y, x0.z, x0.w, x1.x, x1.y, x1.z, x1.w};
            float ms[8] = {m0.x, m0.y, m0.z, m0.w, m1.x, m1.y, m1.z, m1.w};
#pragma unroll
            for (int i = 0; i < 8; ++i) hv[i] = (_Float16)(xs[i] * ms[i]);
        } else {
#pragma unroll
            for (int i = 0; i < 8; ++i) hv[i] = (_Float16)0.0f;
        }
        *(v8h*)&aTile[e] = hv;
    }
    __syncthreads();

    const int wave    = tid >> 5;
    const int lane    = tid & 31;
    const int laneRow = lane & 15;       // A row / B col / C col
    const int laneHi  = lane >> 4;       // half-wave selector
    const int col0    = wave * 16;

    v8f acc = {};

#pragma unroll
    for (int ks = 0; ks < 4; ++ks) {     // K steps of 32
        // ---- A fragment: row = laneRow; chunks K=[kA0,kA0+8) and K=[kA0+16,kA0+24)
        int kA0 = ks * 32 + laneHi * 8;
        v8h lo = *(const v8h*)&aTile[laneRow * D + kA0];
        v8h hi = *(const v8h*)&aTile[laneRow * D + kA0 + 16];
        v16h a;
#pragma unroll
        for (int i = 0; i < 8; ++i) { a[i] = lo[i]; a[i + 8] = hi[i]; }

        // ---- B fragment: col n = col0+laneRow; chunks K=[kB,kB+8) and K=[kB+16,kB+24)
        int n  = col0 + laneRow;
        int kB = ks * 32 + laneHi * 8;
        v8h blo = *(const v8h*)(wT + (size_t)n * D + kB);
        v8h bhi = *(const v8h*)(wT + (size_t)n * D + kB + 16);
        v16h b;
#pragma unroll
        for (int i = 0; i < 8; ++i) { b[i] = blo[i]; b[i + 8] = bhi[i]; }

        acc = __builtin_amdgcn_wmma_f32_16x16x32_f16(
            /*neg_a=*/false, a, /*neg_b=*/false, b,
            /*c_mod=*/(short)0, acc, /*reuse_a=*/false, /*reuse_b=*/false);
    }

    // ---- transpose-free epilogue through LDS for coalesced global stores ----
    // C layout: VGPR r -> local row = r + 8*laneHi, col = col0 + laneRow
    __syncthreads();                     // all waves done reading aTile
    {
        const int lr = 8 * laneHi;
        const int n  = col0 + laneRow;
#pragma unroll
        for (int r = 0; r < 8; ++r)
            aTile[(lr + r) * D + n] = (_Float16)acc[r];
    }
    __syncthreads();
    {
        int e = tid * 8;
        int r = e >> 7;
        int k = e & (D - 1);
        int row = row0 + r;
        if (row < nNodes)
            *(v8h*)(h16 + (size_t)row * D + k) = *(const v8h*)&aTile[e];
    }
}

// ---------------------------------------------------------------------------
// Kernel 2: zero the accumulator (d_out)
// ---------------------------------------------------------------------------
__global__ __launch_bounds__(256) void zero_f4_kernel(float4* __restrict__ p, int n4) {
    int i = blockIdx.x * blockDim.x + threadIdx.x;
    if (i < n4) p[i] = make_float4(0.f, 0.f, 0.f, 0.f);
}

// ---------------------------------------------------------------------------
// Kernel 3: edge scatter. One wave per edge; lane covers 4 features.
// h16 (25.6 MB) and out (51.2 MB) both L2-resident; fp32 atomics at L2.
// ---------------------------------------------------------------------------
__global__ __launch_bounds__(256) void scatter_kernel(const _Float16* __restrict__ h,
                                                      const int* __restrict__ erow,
                                                      const int* __restrict__ ecol,
                                                      const float* __restrict__ eval_,
                                                      float* __restrict__ out,
                                                      int nEdges) {
    const int lane = threadIdx.x & 31;
    const int e = blockIdx.x * 8 + (threadIdx.x >> 5);
    if (e >= nEdges) return;

    const int   src = ecol[e];
    const int   dst = erow[e];
    const float v   = eval_[e];

    v4h hv = *(const v4h*)(h + (size_t)src * D + lane * 4);
    float* op = out + (size_t)dst * D + lane * 4;

#pragma unroll
    for (int j = 0; j < 4; ++j) {
        float m = (float)hv[j] * v;
        float* p = op + j;
        // non-returning device-scope fp32 atomic add (hardware atomic, no CAS loop)
        asm volatile("global_atomic_add_f32 %0, %1, off scope:SCOPE_DEV"
                     :: "v"(p), "v"(m) : "memory");
    }
}

// ---------------------------------------------------------------------------
// Kernel 4: out = relu(out + b), float4-vectorized
// ---------------------------------------------------------------------------
__global__ __launch_bounds__(256) void bias_relu_kernel(float* __restrict__ out,
                                                        const float* __restrict__ b,
                                                        int n4) {
    int i = blockIdx.x * blockDim.x + threadIdx.x;
    if (i >= n4) return;
    float4 o  = ((float4*)out)[i];
    float4 bb = ((const float4*)b)[i & (D4 - 1)];
    o.x = fmaxf(o.x + bb.x, 0.f);
    o.y = fmaxf(o.y + bb.y, 0.f);
    o.z = fmaxf(o.z + bb.z, 0.f);
    o.w = fmaxf(o.w + bb.w, 0.f);
    ((float4*)out)[i] = o;
}

// ---------------------------------------------------------------------------
extern "C" void kernel_launch(void* const* d_in, const int* in_sizes, int n_in,
                              void* d_out, int out_size, void* d_ws, size_t ws_size,
                              hipStream_t stream) {
    const float* x     = (const float*)d_in[0];
    const float* w     = (const float*)d_in[1];
    const float* b     = (const float*)d_in[2];
    const int*   erow  = (const int*)d_in[3];
    const int*   ecol  = (const int*)d_in[4];
    const float* eval_ = (const float*)d_in[5];
    const float* mask  = (const float*)d_in[6];

    const int nNodes = in_sizes[0] / D;
    const int nEdges = in_sizes[3];

    // workspace layout: [wT f16 128x128 = 32 KB][h16 f16 nNodes x 128]
    _Float16* wT  = (_Float16*)d_ws;
    _Float16* h16 = (_Float16*)((char*)d_ws + (size_t)D * D * sizeof(_Float16));
    float*    out = (float*)d_out;

    // 0) w -> wT (f16, transposed)
    prep_w_kernel<<<(D * D) / 256, 256, 0, stream>>>(w, wT);

    // 1) fused dropout + WMMA GEMM -> h16
    int rowTiles = (nNodes + 15) / 16;
    gemm_wmma_kernel<<<rowTiles, 256, 0, stream>>>(x, mask, wT, h16, nNodes);

    // 2) zero accumulator
    int n4 = nNodes * D4;
    zero_f4_kernel<<<(n4 + 255) / 256, 256, 0, stream>>>((float4*)out, n4);

    // 3) edge scatter with device-scope fp32 atomics
    scatter_kernel<<<(nEdges + 7) / 8, 256, 0, stream>>>(h16, erow, ecol, eval_, out, nEdges);

    // 4) bias + relu
    bias_relu_kernel<<<(n4 + 255) / 256, 256, 0, stream>>>(out, b, n4);
}
